// SimpleGNN_26869315404005
// MI455X (gfx1250) — matrix operand order repaired and checked by reading, not compiled, not run on hardware
//
#include <hip/hip_runtime.h>
#include <hip/hip_bf16.h>

#define N_NODES  20000
#define N_EDGES  300000
#define N_GRAPHS 64
#define F_IN     128
#define HIDDEN   256

typedef __attribute__((ext_vector_type(16))) _Float16 v16h;
typedef __attribute__((ext_vector_type(8)))  float    v8f;

// ---------------- utility fills (graph-capture safe, no hipMemset) ----------
__global__ void k_fill_f32(float* __restrict__ p, float v, int n) {
  int i = blockIdx.x * blockDim.x + threadIdx.x;
  if (i < n) p[i] = v;
}
__global__ void k_fill_u32(unsigned* __restrict__ p, unsigned v, int n) {
  int i = blockIdx.x * blockDim.x + threadIdx.x;
  if (i < n) p[i] = v;
}

// ---------------- degree / normalization ------------------------------------
__global__ void k_count_deg(const int* __restrict__ dst, float* __restrict__ deg) {
  int i = blockIdx.x * blockDim.x + threadIdx.x;
  if (i < N_EDGES) atomicAdd(&deg[dst[i]], 1.0f);
}
__global__ void k_dinv(float* __restrict__ d) {  // deg -> rsqrt(deg + 1)
  int i = blockIdx.x * blockDim.x + threadIdx.x;
  if (i < N_NODES) d[i] = rsqrtf(d[i] + 1.0f);
}

// ---------------- fragment loaders (ISA 7.12.2 layouts) ----------------------
// A-matrix 16x32 f16 fragment for rows [rowBase, rowBase+16)
__device__ __forceinline__ v16h load_a_frag(const float* __restrict__ A, int row,
                                            int K, int k0, int khalf) {
  v16h a;
#pragma unroll
  for (int v = 0; v < 8; ++v) {
    const int ka = k0 + ((v >> 2) << 4) + (khalf << 3) + ((v & 3) << 1);
    a[2 * v]     = (_Float16)A[row * K + ka];
    a[2 * v + 1] = (_Float16)A[row * K + ka + 1];
  }
  return a;
}
// B-matrix 32x16 f16 fragment for column `col`
__device__ __forceinline__ v16h load_b_frag(const float* __restrict__ W, int col,
                                            int Nout, int k0, int khalf) {
  v16h b;
#pragma unroll
  for (int v = 0; v < 8; ++v) {
    const int kb = k0 + ((v >> 2) << 4) + (khalf << 3) + ((v & 3) << 1);
    b[2 * v]     = (_Float16)W[kb * Nout + col];
    b[2 * v + 1] = (_Float16)W[(kb + 1) * Nout + col];
  }
  return b;
}

__device__ __forceinline__ v8f wmma_f16(v16h a, v16h b, v8f c) {
  return __builtin_amdgcn_wmma_f32_16x16x32_f16(false, a, false, b, (short)0, c,
                                                false, false);
}

// ---------------- WMMA GEMM: C[M,Nout] = op(A[M,K] @ W[K,Nout] (+bias)(relu))
// One wave32 per 32x32 output block (2x2 register tiling: 4 accumulators,
// A/B fragments shared across two WMMAs each). fp32 in, f16 operands, f32 acc.
// flags: bit0 = add bias, bit1 = relu
__launch_bounds__(256)
__global__ void k_gemm_wmma(const float* __restrict__ A, const float* __restrict__ W,
                            const float* __restrict__ bias, float* __restrict__ C,
                            int M, int K, int Nout, int flags) {
  const int lane   = threadIdx.x & 31;
  const int wave   = threadIdx.x >> 5;
  const int strip  = blockIdx.x * 8 + wave;          // 8 waves per block
  const int stripsN = Nout >> 5;                      // 32-wide column strips
  const int stripsM = M >> 5;
  if (strip >= stripsM * stripsN) return;             // wave-uniform exit
  const int sm = strip / stripsN;
  const int sn = strip - sm * stripsN;
  const int mlane = lane & 15;
  const int khalf = lane >> 4;
  const int arow0 = sm * 32 + mlane;                  // A rows for this lane
  const int arow1 = arow0 + 16;
  const int col0  = sn * 32 + mlane;                  // B/C columns for this lane
  const int col1  = col0 + 16;

  v8f acc00 = {}, acc01 = {}, acc10 = {}, acc11 = {};
  for (int k0 = 0; k0 < K; k0 += 32) {
    const v16h a0 = load_a_frag(A, arow0, K, k0, khalf);
    const v16h a1 = load_a_frag(A, arow1, K, k0, khalf);
    const v16h b0 = load_b_frag(W, col0, Nout, k0, khalf);
    const v16h b1 = load_b_frag(W, col1, Nout, k0, khalf);
    acc00 = wmma_f16(a0, b0, acc00);
    acc01 = wmma_f16(a0, b1, acc01);
    acc10 = wmma_f16(a1, b0, acc10);
    acc11 = wmma_f16(a1, b1, acc11);
  }

  const float bb0 = (flags & 1) ? bias[col0] : 0.0f;
  const float bb1 = (flags & 1) ? bias[col1] : 0.0f;
  const int mbase0 = sm * 32 + (khalf << 3);          // C/D layout: M = v + 8*khalf
  const int mbase1 = mbase0 + 16;
#pragma unroll
  for (int v = 0; v < 8; ++v) {
    float v00 = acc00[v] + bb0, v01 = acc01[v] + bb1;
    float v10 = acc10[v] + bb0, v11 = acc11[v] + bb1;
    if (flags & 2) {
      v00 = fmaxf(v00, 0.0f); v01 = fmaxf(v01, 0.0f);
      v10 = fmaxf(v10, 0.0f); v11 = fmaxf(v11, 0.0f);
    }
    C[(mbase0 + v) * Nout + col0] = v00;
    C[(mbase0 + v) * Nout + col1] = v01;
    C[(mbase1 + v) * Nout + col0] = v10;
    C[(mbase1 + v) * Nout + col1] = v11;
  }
}

// ---------------- edge scatter-add: agg[dst] += h[src] * dinv[src]*dinv[dst] -
__global__ void k_aggregate(const float* __restrict__ h, const int* __restrict__ src,
                            const int* __restrict__ dst, const float* __restrict__ dinv,
                            float* __restrict__ agg) {
  const int e = blockIdx.x;
  const int s = src[e], d = dst[e];
  const float norm = dinv[s] * dinv[d];
  const int f = threadIdx.x;  // HIDDEN == blockDim
  atomicAdd(&agg[d * HIDDEN + f], h[s * HIDDEN + f] * norm);
}

// ---------------- agg = relu(agg + h*dinv^2 + b) -----------------------------
__global__ void k_finalize(float* __restrict__ agg, const float* __restrict__ h,
                           const float* __restrict__ dinv, const float* __restrict__ b) {
  int idx = blockIdx.x * blockDim.x + threadIdx.x;
  if (idx < N_NODES * HIDDEN) {
    const int i = idx >> 8;        // / HIDDEN (256)
    const int f = idx & 255;
    const float di = dinv[i];
    agg[idx] = fmaxf(agg[idx] + h[idx] * di * di + b[f], 0.0f);
  }
}

// ---------------- gate = G1 @ gW2 + gb2 -------------------------------------
__global__ void k_gate(const float* __restrict__ G1, const float* __restrict__ gW2,
                       const float* __restrict__ gb2, float* __restrict__ gate) {
  int n = blockIdx.x * blockDim.x + threadIdx.x;
  if (n < N_NODES) {
    float acc = gb2[0];
#pragma unroll 4
    for (int j = 0; j < 128; ++j) acc += G1[n * 128 + j] * gW2[j];
    gate[n] = acc;
  }
}

// ---------------- ordered-float encoding for atomic max ----------------------
__device__ __forceinline__ unsigned ford(float f) {
  unsigned u = __float_as_uint(f);
  return (u & 0x80000000u) ? ~u : (u | 0x80000000u);
}
__device__ __forceinline__ float dord(unsigned u) {
  return (u & 0x80000000u) ? __uint_as_float(u & 0x7FFFFFFFu) : __uint_as_float(~u);
}

__global__ void k_segmax(const float* __restrict__ gate, const int* __restrict__ batch,
                         unsigned* __restrict__ menc) {
  int n = blockIdx.x * blockDim.x + threadIdx.x;
  if (n < N_NODES) atomicMax(&menc[batch[n]], ford(gate[n]));
}

__global__ void k_expsum(const float* __restrict__ gate, const int* __restrict__ batch,
                         const unsigned* __restrict__ menc, float* __restrict__ eexp,
                         float* __restrict__ ssum) {
  int n = blockIdx.x * blockDim.x + threadIdx.x;
  if (n < N_NODES) {
    const int g = batch[n];
    const float e = expf(gate[n] - dord(menc[g]));
    eexp[n] = e;
    atomicAdd(&ssum[g], e);
  }
}

__global__ void k_pool(const float* __restrict__ h2, const int* __restrict__ batch,
                       const float* __restrict__ eexp, const float* __restrict__ ssum,
                       float* __restrict__ pooled) {
  const int n = blockIdx.x;
  const int f = threadIdx.x;                // HIDDEN threads
  const int g = batch[n];
  const float a = eexp[n] / ssum[g];
  atomicAdd(&pooled[g * HIDDEN + f], a * h2[n * HIDDEN + f]);
}

// ---------------- classifier head: out = relu(p@cW1+cb1)@cW2+cb2 ------------
__global__ void k_classifier(const float* __restrict__ pooled, const float* __restrict__ cW1,
                             const float* __restrict__ cb1, const float* __restrict__ cW2,
                             const float* __restrict__ cb2, float* __restrict__ out) {
  const int g = blockIdx.x;
  const int j = threadIdx.x;                // 128 threads
  float t = cb1[j];
  for (int k = 0; k < HIDDEN; ++k) t += pooled[g * HIDDEN + k] * cW1[k * 128 + j];
  t = fmaxf(t, 0.0f);
  __shared__ float r0[128], r1[128];
  r0[j] = t * cW2[j * 2 + 0];
  r1[j] = t * cW2[j * 2 + 1];
  __syncthreads();
  for (int s = 64; s > 0; s >>= 1) {
    if (j < s) { r0[j] += r0[j + s]; r1[j] += r1[j + s]; }
    __syncthreads();
  }
  if (j == 0) {
    out[g * 2 + 0] = r0[0] + cb2[0];
    out[g * 2 + 1] = r1[0] + cb2[1];
  }
}

// ---------------------------------------------------------------------------
extern "C" void kernel_launch(void* const* d_in, const int* in_sizes, int n_in,
                              void* d_out, int out_size, void* d_ws, size_t ws_size,
                              hipStream_t stream) {
  (void)in_sizes; (void)n_in; (void)out_size; (void)ws_size;
  const float* x    = (const float*)d_in[0];
  const int*   ei   = (const int*)d_in[1];
  const int*   batch= (const int*)d_in[2];
  const float* W1   = (const float*)d_in[3];
  const float* b1   = (const float*)d_in[4];
  const float* W2   = (const float*)d_in[5];
  const float* b2   = (const float*)d_in[6];
  const float* gW1  = (const float*)d_in[7];
  const float* gb1  = (const float*)d_in[8];
  const float* gW2  = (const float*)d_in[9];
  const float* gb2  = (const float*)d_in[10];
  const float* cW1  = (const float*)d_in[11];
  const float* cb1  = (const float*)d_in[12];
  const float* cW2  = (const float*)d_in[13];
  const float* cb2  = (const float*)d_in[14];
  float* out = (float*)d_out;
  const int* src = ei;
  const int* dst = ei + N_EDGES;

  // workspace carving (256B aligned)
  char* base = (char*)d_ws;
  size_t off = 0;
  auto carve = [&](size_t bytes) -> char* {
    char* p = base + off;
    off += (bytes + 255) & ~(size_t)255;
    return p;
  };
  float*    RA     = (float*)carve((size_t)N_NODES * HIDDEN * sizeof(float));
  float*    RB     = (float*)carve((size_t)N_NODES * HIDDEN * sizeof(float));
  float*    dinv   = (float*)carve((size_t)N_NODES * sizeof(float));
  float*    gate   = (float*)carve((size_t)N_NODES * sizeof(float));
  float*    eexp   = (float*)carve((size_t)N_NODES * sizeof(float));
  unsigned* menc   = (unsigned*)carve((size_t)N_GRAPHS * sizeof(unsigned));
  float*    ssum   = (float*)carve((size_t)N_GRAPHS * sizeof(float));
  float*    pooled = (float*)carve((size_t)N_GRAPHS * HIDDEN * sizeof(float));

  const int T = 256;
  const int gNodes = (N_NODES + T - 1) / T;
  const int gNH    = (N_NODES * HIDDEN + T - 1) / T;
  auto gemm_blocks = [](int M, int Nout) {
    const int strips = (M >> 5) * (Nout >> 5);
    return (strips + 7) / 8;                     // 8 waves per block
  };

  // degrees -> dinv
  k_fill_f32<<<gNodes, T, 0, stream>>>(dinv, 0.0f, N_NODES);
  k_count_deg<<<(N_EDGES + T - 1) / T, T, 0, stream>>>(dst, dinv);
  k_dinv<<<gNodes, T, 0, stream>>>(dinv);

  // layer 1: H = X @ W1 -> RA; agg -> RB; RB = relu(RB + RA*dinv^2 + b1)
  k_gemm_wmma<<<gemm_blocks(N_NODES, HIDDEN), 256, 0, stream>>>(
      x, W1, nullptr, RA, N_NODES, F_IN, HIDDEN, 0);
  k_fill_f32<<<gNH, T, 0, stream>>>(RB, 0.0f, N_NODES * HIDDEN);
  k_aggregate<<<N_EDGES, HIDDEN, 0, stream>>>(RA, src, dst, dinv, RB);
  k_finalize<<<gNH, T, 0, stream>>>(RB, RA, dinv, b1);

  // layer 2: H2 = RB @ W2 -> RA; agg -> RB; RB = relu(...)  (RB = node embeddings)
  k_gemm_wmma<<<gemm_blocks(N_NODES, HIDDEN), 256, 0, stream>>>(
      RB, W2, nullptr, RA, N_NODES, HIDDEN, HIDDEN, 0);
  k_fill_f32<<<gNH, T, 0, stream>>>(RB, 0.0f, N_NODES * HIDDEN);
  k_aggregate<<<N_EDGES, HIDDEN, 0, stream>>>(RA, src, dst, dinv, RB);
  k_finalize<<<gNH, T, 0, stream>>>(RB, RA, dinv, b2);

  // gate MLP: G1 = relu(RB @ gW1 + gb1) -> RA [N,128]; gate = G1 @ gW2 + gb2
  k_gemm_wmma<<<gemm_blocks(N_NODES, 128), 256, 0, stream>>>(
      RB, gW1, gb1, RA, N_NODES, HIDDEN, 128, 3);
  k_gate<<<gNodes, T, 0, stream>>>(RA, gW2, gb2, gate);

  // segment softmax + weighted pooling
  k_fill_u32<<<1, N_GRAPHS, 0, stream>>>(menc, 0u, N_GRAPHS);
  k_fill_f32<<<1, N_GRAPHS, 0, stream>>>(ssum, 0.0f, N_GRAPHS);
  k_fill_f32<<<(N_GRAPHS * HIDDEN + T - 1) / T, T, 0, stream>>>(pooled, 0.0f, N_GRAPHS * HIDDEN);
  k_segmax<<<gNodes, T, 0, stream>>>(gate, batch, menc);
  k_expsum<<<gNodes, T, 0, stream>>>(gate, batch, menc, eexp, ssum);
  k_pool<<<N_NODES, HIDDEN, 0, stream>>>(RB, batch, eexp, ssum, pooled);

  // classifier head -> [64, 2]
  k_classifier<<<N_GRAPHS, 128, 0, stream>>>(pooled, cW1, cb1, cW2, cb2, out);
}
